// RigidDiffusion_60541859004915
// MI455X (gfx1250) — matrix operand order repaired and checked by reading, not compiled, not run on hardware
//
#include <hip/hip_runtime.h>

// Problem constants (match reference)
#define BB    2
#define NN    1280
#define KK    48
#define BNrows (BB*NN)          // 2560
#define CNODE 384
#define CEDGE 128
#define CBIAS 192
#define HID   512
#define COUT  128
#define MROWS (BB*NN*KK)        // 122880
#define LN_EPS 1e-5f

typedef __attribute__((ext_vector_type(16))) _Float16 v16h;
typedef __attribute__((ext_vector_type(8)))  _Float16 v8h;
typedef __attribute__((ext_vector_type(4)))  _Float16 v4h;
typedef __attribute__((ext_vector_type(8)))  float    v8f;
typedef __attribute__((ext_vector_type(4)))  float    v4f;

// ---------------------------------------------------------------------------
// Prolog 1: n = node_emb @ Wi + bi  (fp32 accumulate, stored f16)
// ---------------------------------------------------------------------------
__global__ void k_embed(const float* __restrict__ node,
                        const float* __restrict__ Wi,
                        const float* __restrict__ bi,
                        _Float16* __restrict__ n16) {
  int idx = blockIdx.x * blockDim.x + threadIdx.x;
  if (idx >= BNrows * CBIAS) return;
  int row = idx / CBIAS;
  int c   = idx - row * CBIAS;
  const float* nr = node + (size_t)row * CNODE;
  float acc = bi[c];
#pragma unroll 8
  for (int k = 0; k < CNODE; ++k)
    acc = fmaf(nr[k], Wi[k * CBIAS + c], acc);
  n16[idx] = (_Float16)acc;
}

// ---------------------------------------------------------------------------
// Prolog 2: pack a [Kt x Nt] fp32 weight into WMMA B-operand f16 layout.
// Packed layout: [(kb*NB+nb)][lane(32)][j(16)]  where for
// V_WMMA_F32_16X16X32_F16 the B matrix (32x16, KxN) is packed here as:
//   lane = 16*hi + n  (n = lane&15),  half j holds K = 16*hi + j
// Each lane reads its v16h fragment as one contiguous 32B chunk.
// ---------------------------------------------------------------------------
__global__ void k_pack(const float* __restrict__ W,
                       _Float16* __restrict__ P, int Kt, int Nt) {
  int idx = blockIdx.x * blockDim.x + threadIdx.x;
  if (idx >= Kt * Nt) return;
  int j    = idx & 15;
  int lane = (idx >> 4) & 31;
  int blk  = idx >> 9;           // kb*NB + nb
  int NB   = Nt >> 4;
  int nb   = blk % NB;
  int kb   = blk / NB;
  int k    = kb * 32 + 16 * (lane >> 4) + j;
  int n    = nb * 16 + (lane & 15);
  P[idx] = (_Float16)W[(size_t)k * Nt + n];
}

// ---------------------------------------------------------------------------
// Fragment loaders
// ---------------------------------------------------------------------------
// A operand (16x32 f16, row-major [rows][HID] tile in LDS).
// Layout: half j holds K = 16*hi + j (matches packed B) => one contiguous 16-half run.
__device__ __forceinline__ v16h ldsA(const _Float16* base, int kb, int m, int hi) {
  const _Float16* p = base + m * HID + kb * 32 + 16 * hi;
  v8h lo = *(const v8h*)(p);
  v8h hh = *(const v8h*)(p + 8);
  return __builtin_shufflevector(lo, hh, 0,1,2,3,4,5,6,7,8,9,10,11,12,13,14,15);
}

__device__ __forceinline__ v16h ldB(const _Float16* __restrict__ P, int blk, int lane) {
  return *(const v16h*)(P + ((size_t)(blk * 32 + lane) << 4));
}

__device__ __forceinline__ v8f wmma_f16(v16h a, v16h b, v8f c) {
  return __builtin_amdgcn_wmma_f32_16x16x32_f16(false, a, false, b, (short)0, c,
                                                false, false);
}

// ---------------------------------------------------------------------------
// Fused trunk: build x -> L1(relu) -> L2(relu)+residual -> final -> LayerNorm
// 2 waves/block; each wave privately owns a 32-row strip (2 WMMA row-blocks)
// so every B fragment feeds 2 WMMAs (halved L2/WGP$ B traffic, 4 acc chains).
// ---------------------------------------------------------------------------
__launch_bounds__(64)
__global__ void k_fused(const float* __restrict__ edge,
                        const int*   __restrict__ eidx,
                        const _Float16* __restrict__ n16,
                        const _Float16* __restrict__ W1p, const float* __restrict__ b1,
                        const _Float16* __restrict__ W2p, const float* __restrict__ b2,
                        const _Float16* __restrict__ Wfp, const float* __restrict__ bf,
                        const float* __restrict__ gamma, const float* __restrict__ beta,
                        float* __restrict__ out) {
  __shared__ __align__(16) _Float16 s_x[64 * HID];   // 64 KB
  __shared__ __align__(16) _Float16 s_h[64 * HID];   // 64 KB

  const int lane = threadIdx.x & 31;
  const int wave = threadIdx.x >> 5;
  const int m    = lane & 15;
  const int hi   = lane >> 4;
  const int rowBase = blockIdx.x * 64 + wave * 32;

  _Float16* xw = s_x + wave * 32 * HID;
  _Float16* hw = s_h + wave * 32 * HID;

  // ---- build x strip: [edge(128) | n_self(192) | n_nbr(192)] as f16 ----
  for (int r = 0; r < 32; ++r) {
    int row = rowBase + r;
    int b   = row / (NN * KK);
    int rem = row - b * (NN * KK);
    int i   = rem / KK;
    _Float16* xr = xw + r * HID;

    // edge: 128 f32 -> f16, float4 per lane
    {
      v4f v = ((const v4f*)(edge + (size_t)row * CEDGE))[lane];
      v4h h4 = { (_Float16)v.x, (_Float16)v.y, (_Float16)v.z, (_Float16)v.w };
      ((v4h*)xr)[lane] = h4;
    }
    // n_self / n_nbr: already f16, copy as dwords (96 dwords each -> 3/lane)
    int jn = eidx[row];
    const unsigned* ns = (const unsigned*)(n16 + (size_t)(b * NN + i)  * CBIAS);
    const unsigned* nn = (const unsigned*)(n16 + (size_t)(b * NN + jn) * CBIAS);
    unsigned* dS = (unsigned*)(xr + CEDGE);
    unsigned* dN = (unsigned*)(xr + CEDGE + CBIAS);
#pragma unroll
    for (int t = 0; t < 3; ++t) {
      dS[lane + 32 * t] = ns[lane + 32 * t];
      dN[lane + 32 * t] = nn[lane + 32 * t];
    }
  }
  __syncthreads();

  // ---- layer 1: h = relu(x @ W1 + b1) ----
  for (int nb = 0; nb < 32; nb += 2) {
    v8f acc[2][2] = {};                       // [row-block][col]
    for (int kb = 0; kb < 16; ++kb) {
      v16h bA = ldB(W1p, kb * 32 + nb,     lane);
      v16h bB = ldB(W1p, kb * 32 + nb + 1, lane);
#pragma unroll
      for (int rb = 0; rb < 2; ++rb) {
        v16h a = ldsA(xw + rb * 16 * HID, kb, m, hi);
        acc[rb][0] = wmma_f16(a, bA, acc[rb][0]);
        acc[rb][1] = wmma_f16(a, bB, acc[rb][1]);
      }
    }
    float bias0 = b1[nb * 16 + m];
    float bias1 = b1[(nb + 1) * 16 + m];
#pragma unroll
    for (int rb = 0; rb < 2; ++rb)
#pragma unroll
      for (int r = 0; r < 8; ++r) {
        int row = rb * 16 + r + 8 * hi;
        hw[row * HID + nb * 16 + m]       = (_Float16)fmaxf(acc[rb][0][r] + bias0, 0.f);
        hw[row * HID + (nb + 1) * 16 + m] = (_Float16)fmaxf(acc[rb][1][r] + bias1, 0.f);
      }
  }
  __syncthreads();

  // ---- layer 2: s = relu(h @ W2 + b2) + x   (s overwrites x in place) ----
  for (int nb = 0; nb < 32; nb += 2) {
    v8f acc[2][2] = {};
    for (int kb = 0; kb < 16; ++kb) {
      v16h bA = ldB(W2p, kb * 32 + nb,     lane);
      v16h bB = ldB(W2p, kb * 32 + nb + 1, lane);
#pragma unroll
      for (int rb = 0; rb < 2; ++rb) {
        v16h a = ldsA(hw + rb * 16 * HID, kb, m, hi);
        acc[rb][0] = wmma_f16(a, bA, acc[rb][0]);
        acc[rb][1] = wmma_f16(a, bB, acc[rb][1]);
      }
    }
    float bias0 = b2[nb * 16 + m];
    float bias1 = b2[(nb + 1) * 16 + m];
#pragma unroll
    for (int rb = 0; rb < 2; ++rb)
#pragma unroll
      for (int r = 0; r < 8; ++r) {
        int row  = rb * 16 + r + 8 * hi;
        int col0 = nb * 16 + m, col1 = col0 + 16;
        float s0 = fmaxf(acc[rb][0][r] + bias0, 0.f) + (float)xw[row * HID + col0];
        float s1 = fmaxf(acc[rb][1][r] + bias1, 0.f) + (float)xw[row * HID + col1];
        xw[row * HID + col0] = (_Float16)s0;
        xw[row * HID + col1] = (_Float16)s1;
      }
  }
  __syncthreads();

  // ---- final: y = s @ Wf + bf  (32x128 f32 staged in LDS, reusing hw) ----
  float* yb = (float*)hw;   // 32*128*4 = 16 KB, fits in the 32 KB hw region
  for (int nb = 0; nb < 8; nb += 2) {
    v8f acc[2][2] = {};
    for (int kb = 0; kb < 16; ++kb) {
      v16h bA = ldB(Wfp, kb * 8 + nb,     lane);
      v16h bB = ldB(Wfp, kb * 8 + nb + 1, lane);
#pragma unroll
      for (int rb = 0; rb < 2; ++rb) {
        v16h a = ldsA(xw + rb * 16 * HID, kb, m, hi);
        acc[rb][0] = wmma_f16(a, bA, acc[rb][0]);
        acc[rb][1] = wmma_f16(a, bB, acc[rb][1]);
      }
    }
    float bias0 = bf[nb * 16 + m];
    float bias1 = bf[(nb + 1) * 16 + m];
#pragma unroll
    for (int rb = 0; rb < 2; ++rb)
#pragma unroll
      for (int r = 0; r < 8; ++r) {
        int row = rb * 16 + r + 8 * hi;
        yb[row * COUT + nb * 16 + m]       = acc[rb][0][r] + bias0;
        yb[row * COUT + (nb + 1) * 16 + m] = acc[rb][1][r] + bias1;
      }
  }
  __syncthreads();

  // ---- LayerNorm over COUT=128 per row: one row per lane (32 rows) ----
  float* statp = (float*)xw;   // xw region free after final GEMM
  {
    const float* yr = yb + lane * COUT;
    float s = 0.f, s2 = 0.f;
    for (int c = 0; c < COUT; ++c) { float v = yr[c]; s += v; s2 += v * v; }
    float mu  = s  * (1.f / COUT);
    float var = s2 * (1.f / COUT) - mu * mu;
    statp[lane]      = mu;
    statp[32 + lane] = rsqrtf(var + LN_EPS);
  }
  __syncthreads();

  for (int r = 0; r < 32; ++r) {
    float mu = statp[r];
    float rs = statp[32 + r];
    const float* yr = yb + r * COUT;
    float* orow = out + (size_t)(rowBase + r) * COUT;
#pragma unroll
    for (int t = 0; t < 4; ++t) {
      int c = lane + 32 * t;
      orow[c] = (yr[c] - mu) * rs * gamma[c] + beta[c];
    }
  }
}

// ---------------------------------------------------------------------------
// Launch
// ---------------------------------------------------------------------------
extern "C" void kernel_launch(void* const* d_in, const int* in_sizes, int n_in,
                              void* d_out, int out_size, void* d_ws, size_t ws_size,
                              hipStream_t stream) {
  const float* node  = (const float*)d_in[0];
  const float* edge  = (const float*)d_in[1];
  const int*   eidx  = (const int*)  d_in[2];
  const float* Wi    = (const float*)d_in[3];
  const float* bi    = (const float*)d_in[4];
  const float* W1    = (const float*)d_in[5];
  const float* b1    = (const float*)d_in[6];
  const float* W2    = (const float*)d_in[7];
  const float* b2    = (const float*)d_in[8];
  const float* Wf    = (const float*)d_in[9];
  const float* bf    = (const float*)d_in[10];
  const float* gamma = (const float*)d_in[11];
  const float* beta  = (const float*)d_in[12];
  float* out = (float*)d_out;

  // Workspace layout (f16): n16 | W1p | W2p | Wfp  = ~2.1 MB total
  _Float16* ws  = (_Float16*)d_ws;
  _Float16* n16 = ws;
  _Float16* W1p = n16 + (size_t)BNrows * CBIAS;   // 491520 halves
  _Float16* W2p = W1p + (size_t)HID * HID;        // 262144
  _Float16* Wfp = W2p + (size_t)HID * HID;        // 262144 (+65536)

  {
    int tot = BNrows * CBIAS;
    k_embed<<<(tot + 255) / 256, 256, 0, stream>>>(node, Wi, bi, n16);
  }
  k_pack<<<(HID * HID + 255) / 256, 256, 0, stream>>>(W1, W1p, HID, HID);
  k_pack<<<(HID * HID + 255) / 256, 256, 0, stream>>>(W2, W2p, HID, HID);
  k_pack<<<(HID * COUT + 255) / 256, 256, 0, stream>>>(Wf, Wfp, HID, COUT);

  k_fused<<<MROWS / 64, 64, 0, stream>>>(edge, eidx, n16,
                                         W1p, b1, W2p, b2, Wfp, bf,
                                         gamma, beta, out);
}
